// GNN_23708219474326
// MI455X (gfx1250) — compile-verified, matrix operand order checked by reading, not run on hardware
//
#include <hip/hip_runtime.h>
#include <hip/hip_bf16.h>

typedef __attribute__((ext_vector_type(16))) _Float16 v16h;
typedef __attribute__((ext_vector_type(8)))  _Float16 v8h;
typedef __attribute__((ext_vector_type(4)))  _Float16 v4h;
typedef __attribute__((ext_vector_type(8)))  float    v8f;

#define DIM 128
#define LDSK 136  // 128 + 8 halves pad: row stride 272B breaks bank aliasing

// ---------------- utility kernels ----------------

__global__ void zero_f32_kernel(float* __restrict__ p, long long n4) {
  long long i = (long long)blockIdx.x * blockDim.x + threadIdx.x;
  if (i < n4) {
    ((float4*)p)[i] = make_float4(0.f, 0.f, 0.f, 0.f);
  }
}

__global__ void zero_scalar_kernel(float* __restrict__ p, long long n) {
  long long i = (long long)blockIdx.x * blockDim.x + threadIdx.x;
  if (i < n) p[i] = 0.f;
}

// ---------------- degree + norm ----------------

__global__ void degree_kernel(const int* __restrict__ src, const int* __restrict__ dst,
                              unsigned* __restrict__ deg_out, unsigned* __restrict__ deg_in,
                              int n_edges) {
  int e = blockIdx.x * blockDim.x + threadIdx.x;
  if (e < n_edges) {
    atomicAdd(&deg_out[src[e]], 1u);
    atomicAdd(&deg_in[dst[e]], 1u);
  }
}

__global__ void norm_kernel(const unsigned* __restrict__ deg_out,
                            const unsigned* __restrict__ deg_in,
                            float* __restrict__ norm_src, float* __restrict__ norm_dst,
                            int n_nodes) {
  int n = blockIdx.x * blockDim.x + threadIdx.x;
  if (n < n_nodes) {
    unsigned o = deg_out[n]; if (o < 1u) o = 1u;
    unsigned d = deg_in[n];  if (d < 1u) d = 1u;
    norm_src[n] = rsqrtf((float)o);
    norm_dst[n] = rsqrtf((float)d);
  }
}

// ---------------- embedding gather: one wave per node ----------------

__global__ void embed_kernel(const int* __restrict__ feat_id, const float* __restrict__ emb,
                             float* __restrict__ h, int n_nodes) {
  int n = blockIdx.x * (blockDim.x >> 5) + (threadIdx.x >> 5);  // wave-uniform
  int lane = threadIdx.x & 31;
  if (n >= n_nodes) return;
  int fid = __builtin_amdgcn_readfirstlane(feat_id[n]);         // scalar load
  ((float4*)(h + (long long)n * DIM))[lane] =
      ((const float4*)(emb + (long long)fid * DIM))[lane];
}

// ---------------- WMMA GEMM: hw = (h * norm_src) @ W ----------------
// W staged once per block into LDS as f16, transposed [n][k] (padded stride),
// so each B fragment is two ds_load_b128. One wave owns a 16x128 output tile:
// 8 N-tiles x 4 K-blocks = 32 WMMAs. hw is row-padded so stores need no guard.
__global__ __launch_bounds__(128) void gemm_wmma_kernel(
    const float* __restrict__ h, const float* __restrict__ W,
    const float* __restrict__ norm_src, float* __restrict__ hw, int n_nodes) {
  __shared__ _Float16 Wt[DIM * LDSK];

  const int t = threadIdx.x;

  // Cooperative stage: Wt[n][k] = (f16) W[k][n].  Thread t owns column n=t.
  {
    const int n = t & (DIM - 1);
#pragma unroll 4
    for (int k4 = 0; k4 < DIM; k4 += 4) {
      v4h hv;
#pragma unroll
      for (int j = 0; j < 4; ++j)
        hv[j] = (_Float16)W[(k4 + j) * DIM + n];
      *(v4h*)&Wt[n * LDSK + k4] = hv;  // ds_store_b64
    }
  }
  __syncthreads();

  const int wave = t >> 5;
  const int lane = t & 31;
  const int rowbase = (blockIdx.x * 4 + wave) * 16;  // wave-uniform
  if (rowbase >= n_nodes) return;                    // uniform exit: EXEC full

  const int m  = lane & 15;
  const int k0 = (lane >> 4) * 8;
  const int row  = rowbase + m;
  const int rowc = row < n_nodes ? row : (n_nodes - 1);  // clamp loads only
  const float ns = norm_src[rowc];
  const float* hrow = h + (long long)rowc * DIM;

  // Preload all 4 K-blocks of A (scaled, f32 -> f16), via float4 loads.
  v16h Afrag[4];
#pragma unroll
  for (int kb = 0; kb < 4; ++kb) {
    const int base = kb * 32 + k0;
    float4 p0 = *(const float4*)(hrow + base);
    float4 p1 = *(const float4*)(hrow + base + 4);
    float4 p2 = *(const float4*)(hrow + base + 16);
    float4 p3 = *(const float4*)(hrow + base + 20);
    v16h a;
    a[0]  = (_Float16)(p0.x * ns); a[1]  = (_Float16)(p0.y * ns);
    a[2]  = (_Float16)(p0.z * ns); a[3]  = (_Float16)(p0.w * ns);
    a[4]  = (_Float16)(p1.x * ns); a[5]  = (_Float16)(p1.y * ns);
    a[6]  = (_Float16)(p1.z * ns); a[7]  = (_Float16)(p1.w * ns);
    a[8]  = (_Float16)(p2.x * ns); a[9]  = (_Float16)(p2.y * ns);
    a[10] = (_Float16)(p2.z * ns); a[11] = (_Float16)(p2.w * ns);
    a[12] = (_Float16)(p3.x * ns); a[13] = (_Float16)(p3.y * ns);
    a[14] = (_Float16)(p3.z * ns); a[15] = (_Float16)(p3.w * ns);
    Afrag[kb] = a;
  }

  const int ncol_lane = lane & 15;
  const int mstore = rowbase + ((lane >> 4) * 8);

#pragma unroll
  for (int nt = 0; nt < 8; ++nt) {
    v8f acc = {};
    const int ncol = nt * 16 + ncol_lane;
    const _Float16* wrow = &Wt[ncol * LDSK];
#pragma unroll
    for (int kb = 0; kb < 4; ++kb) {
      const int base = kb * 32 + k0;
      v8h b0 = *(const v8h*)(wrow + base);       // ds_load_b128
      v8h b1 = *(const v8h*)(wrow + base + 16);  // ds_load_b128
      v16h B = __builtin_shufflevector(b0, b1, 0, 1, 2, 3, 4, 5, 6, 7,
                                       8, 9, 10, 11, 12, 13, 14, 15);
      acc = __builtin_amdgcn_wmma_f32_16x16x32_f16(false, Afrag[kb], false, B,
                                                   (short)0, acc, false, false);
    }
    // Branchless stores: hw has rows padded to a multiple of 16; rows in
    // [n_nodes, n_pad) are scratch that nothing reads.
    float* hcol = hw + ncol;
#pragma unroll
    for (int r = 0; r < 8; ++r)
      hcol[(long long)(mstore + r) * DIM] = acc[r];
  }
}

// ---------------- edge scatter: agg[dst] += hw[src], one wave per edge ----------------

__global__ void scatter_kernel(const float* __restrict__ hw, const int* __restrict__ src,
                               const int* __restrict__ dst, float* __restrict__ agg,
                               int n_edges) {
  int e = blockIdx.x * (blockDim.x >> 5) + (threadIdx.x >> 5);  // wave-uniform
  int lane = threadIdx.x & 31;
  if (e >= n_edges) return;
  int s = __builtin_amdgcn_readfirstlane(src[e]);  // scalar loads
  int q = __builtin_amdgcn_readfirstlane(dst[e]);
  float4 v = ((const float4*)(hw + (long long)s * DIM))[lane];  // coalesced b128
  float* ap = agg + (long long)q * DIM + lane * 4;
  atomicAdd(ap + 0, v.x);
  atomicAdd(ap + 1, v.y);
  atomicAdd(ap + 2, v.z);
  atomicAdd(ap + 3, v.w);
}

// ---------------- finish: h = relu(agg * norm_dst + b)  (float4) ----------------

__global__ void finish_kernel(const float* __restrict__ agg, const float* __restrict__ norm_dst,
                              const float* __restrict__ bvec, float* __restrict__ h,
                              int n_nodes) {
  long long i = (long long)blockIdx.x * blockDim.x + threadIdx.x;  // over n_nodes*32
  if (i < (long long)n_nodes * (DIM / 4)) {
    int n = (int)(i >> 5);
    int c = (int)(i & 31);
    float nd = norm_dst[n];
    float4 a = ((const float4*)agg)[i];
    float4 bb = ((const float4*)bvec)[c];
    float4 v;
    v.x = fmaxf(a.x * nd + bb.x, 0.f);
    v.y = fmaxf(a.y * nd + bb.y, 0.f);
    v.z = fmaxf(a.z * nd + bb.z, 0.f);
    v.w = fmaxf(a.w * nd + bb.w, 0.f);
    ((float4*)h)[i] = v;
  }
}

// ---------------- readout: per-graph segment max, one wave per node ----------------

__global__ void readout_kernel(const float* __restrict__ h, const int* __restrict__ graph_ids,
                               float* __restrict__ out, int n_nodes) {
  int n = blockIdx.x * (blockDim.x >> 5) + (threadIdx.x >> 5);  // wave-uniform
  int lane = threadIdx.x & 31;
  if (n >= n_nodes) return;
  int g = __builtin_amdgcn_readfirstlane(graph_ids[n]);  // scalar load
  float4 v = ((const float4*)(h + (long long)n * DIM))[lane];
  // h >= 0 post-ReLU, so IEEE bit pattern is order-preserving as unsigned.
  unsigned* op = (unsigned*)(out + (long long)g * DIM + lane * 4);
  atomicMax(op + 0, __float_as_uint(v.x));
  atomicMax(op + 1, __float_as_uint(v.y));
  atomicMax(op + 2, __float_as_uint(v.z));
  atomicMax(op + 3, __float_as_uint(v.w));
}

// ---------------- launch ----------------

extern "C" void kernel_launch(void* const* d_in, const int* in_sizes, int n_in,
                              void* d_out, int out_size, void* d_ws, size_t ws_size,
                              hipStream_t stream) {
  const int* feat_id   = (const int*)d_in[0];
  const int* src       = (const int*)d_in[1];
  const int* dst       = (const int*)d_in[2];
  const int* graph_ids = (const int*)d_in[3];
  const float* emb     = (const float*)d_in[4];
  const float* W       = (const float*)d_in[5];
  const float* bvec    = (const float*)d_in[6];
  float* out = (float*)d_out;

  const int n_nodes = in_sizes[0];
  const int n_edges = in_sizes[1];
  const int layers  = in_sizes[5] / (DIM * DIM);
  const int n_pad   = (n_nodes + 15) & ~15;  // hw rows padded to tile multiple

  // workspace layout (256B aligned)
  char* ws = (char*)d_ws;
  size_t off = 0;
  auto take = [&](size_t bytes) {
    size_t o = off;
    off = (off + bytes + 255) & ~(size_t)255;
    return o;
  };
  float*    h        = (float*)(ws + take((size_t)n_nodes * DIM * sizeof(float)));
  float*    hw       = (float*)(ws + take((size_t)n_pad * DIM * sizeof(float)));
  float*    agg      = (float*)(ws + take((size_t)n_nodes * DIM * sizeof(float)));
  float*    norm_src = (float*)(ws + take((size_t)n_nodes * sizeof(float)));
  float*    norm_dst = (float*)(ws + take((size_t)n_nodes * sizeof(float)));
  unsigned* deg_out  = (unsigned*)(ws + take((size_t)2 * n_nodes * sizeof(unsigned)));
  unsigned* deg_in   = deg_out + n_nodes;
  (void)ws_size;

  const int TPB = 256;
  const int WPB = TPB / 32;  // waves per block
  const long long nd  = (long long)n_nodes * DIM;
  const long long nd4 = nd / 4;
  dim3 blk(TPB);

  // degrees -> norms
  {
    long long n2 = 2LL * n_nodes;
    zero_scalar_kernel<<<(unsigned)((n2 + TPB - 1) / TPB), blk, 0, stream>>>((float*)deg_out, n2);
    degree_kernel<<<(unsigned)((n_edges + TPB - 1) / TPB), blk, 0, stream>>>(src, dst, deg_out, deg_in, n_edges);
    norm_kernel<<<(unsigned)((n_nodes + TPB - 1) / TPB), blk, 0, stream>>>(deg_out, deg_in, norm_src, norm_dst, n_nodes);
  }

  // embedding gather (wave per node)
  embed_kernel<<<(unsigned)((n_nodes + WPB - 1) / WPB), blk, 0, stream>>>(feat_id, emb, h, n_nodes);

  // layers
  const unsigned gemm_blocks = (unsigned)((n_nodes + 63) / 64);  // 4 waves x 16 rows
  for (int l = 0; l < layers; ++l) {
    gemm_wmma_kernel<<<gemm_blocks, dim3(128), 0, stream>>>(h, W + (long long)l * DIM * DIM,
                                                            norm_src, hw, n_nodes);
    zero_f32_kernel<<<(unsigned)((nd4 + TPB - 1) / TPB), blk, 0, stream>>>(agg, nd4);
    scatter_kernel<<<(unsigned)((n_edges + WPB - 1) / WPB), blk, 0, stream>>>(
        hw, src, dst, agg, n_edges);
    finish_kernel<<<(unsigned)((nd4 + TPB - 1) / TPB), blk, 0, stream>>>(
        agg, norm_dst, bvec + (long long)l * DIM, h, n_nodes);
  }

  // readout: zero output then bitwise atomic max (values are >= 0)
  zero_scalar_kernel<<<(unsigned)((out_size + TPB - 1) / TPB), blk, 0, stream>>>(out, out_size);
  readout_kernel<<<(unsigned)((n_nodes + WPB - 1) / WPB), blk, 0, stream>>>(h, graph_ids, out, n_nodes);
}
